// LUConv_76441827934576
// MI455X (gfx1250) — compile-verified, matrix operand order checked by reading, not run on hardware
//
#include <hip/hip_runtime.h>
#include <hip/hip_bf16.h>

typedef __attribute__((ext_vector_type(16))) __bf16 v16bf;
typedef __attribute__((ext_vector_type(8)))  __bf16 v8bf;
typedef __attribute__((ext_vector_type(8)))  float  v8f;

// Types matching __builtin_amdgcn_global_load_async_to_lds_b128's parameters:
// (int __vector_size__(16) in AS1, same in AS3, imm offset, imm cpol)
typedef int int4v __attribute__((vector_size(16)));
typedef __attribute__((address_space(1))) int4v gint4;
typedef __attribute__((address_space(3))) int4v lint4;

union V16U { v16bf v; v8bf h[2]; };

#if __has_builtin(__builtin_amdgcn_global_load_async_to_lds_b128)
#define HAVE_ASYNC_LDS 1
#else
#define HAVE_ASYNC_LDS 0
#endif

__device__ __forceinline__ void wait_async0() {
#if __has_builtin(__builtin_amdgcn_s_wait_asynccnt)
  __builtin_amdgcn_s_wait_asynccnt(0);
#else
  asm volatile("s_wait_asynccnt 0x0" ::: "memory");
#endif
}

__device__ __forceinline__ __bf16 f2bf(float f) {
  unsigned u; __builtin_memcpy(&u, &f, 4);
  u = (u + 0x7FFFu + ((u >> 16) & 1u)) >> 16;     // round-to-nearest-even
  unsigned short s = (unsigned short)u;
  __bf16 r; __builtin_memcpy(&r, &s, 2);
  return r;
}

// ---------------- hash map build ----------------
__global__ void fill_map_kernel(int* __restrict__ map, int n) {
  int i = blockIdx.x * 256 + threadIdx.x;
  if (i < n) map[i] = -1;
}

__global__ void scatter_map_kernel(const int* __restrict__ coords, int* __restrict__ map, int n) {
  int i = blockIdx.x * 256 + threadIdx.x;
  if (i < n) {
    int z = coords[3 * i + 0], y = coords[3 * i + 1], x = coords[3 * i + 2];
    map[(z * 128 + y) * 128 + x] = i;
  }
}

__global__ void zero_page_kernel(unsigned* __restrict__ p) { p[threadIdx.x] = 0u; }  // 32 thr -> 128B

// ---------------- feats f32 -> bf16 (8 elems / thread) ----------------
__global__ void feats_to_bf16_kernel(const float* __restrict__ f, __bf16* __restrict__ o, int n) {
  int i = (blockIdx.x * 256 + threadIdx.x) * 8;
  if (i < n) {
    v8bf r;
#pragma unroll
    for (int j = 0; j < 8; ++j) r[j] = f2bf(f[i + j]);
    *(v8bf*)(o + i) = r;
  }
}

// ---------------- pack Wt [125][Cin=64][Cout=64] into WMMA-B lane layout ----------------
// Wp element index = (((k*2 + kchunk)*4 + ntile)*32 + lane)*16 + i
// lane 0-15 : K = 32*kchunk + {0..7 (i<8), 16..23 (i>=8)}
// lane 16-31: K = 32*kchunk + {8..15, 24..31}
__global__ void pack_w_kernel(const float* __restrict__ Wt, __bf16* __restrict__ Wp) {
  int idx = blockIdx.x * 256 + threadIdx.x;
  if (idx >= 125 * 4096) return;
  int i    =  idx        & 15;
  int lane = (idx >> 4)  & 31;
  int nt   = (idx >> 9)  & 3;
  int c    = (idx >> 11) & 1;
  int k    =  idx >> 12;
  int cin  = 32 * c + (i < 8 ? i : i + 8) + (lane < 16 ? 0 : 8);
  int cout = nt * 16 + (lane & 15);
  Wp[idx] = f2bf(Wt[((size_t)k * 64 + cin) * 64 + cout]);
}

// ---------------- submanifold conv: 128-voxel tile per WG, 8 waves, WMMA bf16 ----------------
// Double-buffered LDS; async global->LDS gather pipelined one offset ahead.
__global__ __launch_bounds__(256) void subm_conv_wmma_kernel(
    const int* __restrict__ coords, const int* __restrict__ map,
    const __bf16* __restrict__ featsb, const __bf16* __restrict__ Wp,
    const float* __restrict__ bias, const __bf16* __restrict__ zpage,
    float* __restrict__ h, int n)
{
  // 2 buffers x 128 rows x 64 ch bf16, row stride 72 elems (144B) -> conflict-free ds_load_b128
  __shared__ __align__(16) __bf16 smem[2][128 * 72];

  const int tid  = threadIdx.x;
  const int lane = tid & 31;
  const int wave = tid >> 5;
  const int r    = tid >> 1;        // gather role: row within tile (2 threads/row)
  const int half = tid & 1;         // gather role: 32-channel half

  const int row = blockIdx.x * 128 + r;
  const bool rowvalid = (row < n);
  int cz = 0, cy = 0, cx = 0;
  if (rowvalid) {
    cz = coords[3 * row + 0];
    cy = coords[3 * row + 1];
    cx = coords[3 * row + 2];
  }

  const int rowInTile = wave * 16 + (lane & 15);   // compute role: A-matrix row
  const int hsel = (lane >= 16) ? 8 : 0;           // K-half select per A layout

  // neighbor row lookup for kernel offset k
  auto lookup = [&](int k) -> int {
    const int kz = k / 25 - 2;
    const int ky = (k / 5) % 5 - 2;
    const int kx = k % 5 - 2;
    int jj = -1;
    if (rowvalid) {
      const int z = cz + kz, y = cy + ky, x = cx + kx;
      if ((unsigned)z < 128u && (unsigned)y < 128u && (unsigned)x < 128u)
        jj = map[(z * 128 + y) * 128 + x];
    }
    return jj;
  };

  // issue this thread's 64B slice of the gather for buffer `buf`
  auto issue = [&](int buf, int jj) {
#if HAVE_ASYNC_LDS
    const __bf16* srcrow = (jj >= 0) ? (featsb + (size_t)jj * 64 + half * 32)
                                     : (zpage + half * 32);
    gint4* gs = (gint4*)(void*)const_cast<__bf16*>(srcrow);
    lint4* ld = (lint4*)(void*)(&smem[buf][r * 72 + half * 32]);
    __builtin_amdgcn_global_load_async_to_lds_b128(gs,     ld,     0, 0);
    __builtin_amdgcn_global_load_async_to_lds_b128(gs + 1, ld + 1, 0, 0);
    __builtin_amdgcn_global_load_async_to_lds_b128(gs + 2, ld + 2, 0, 0);
    __builtin_amdgcn_global_load_async_to_lds_b128(gs + 3, ld + 3, 0, 0);
#else
    uint4* dst = (uint4*)(&smem[buf][r * 72 + half * 32]);
    if (jj >= 0) {
      const uint4* src = (const uint4*)(featsb + (size_t)jj * 64 + half * 32);
#pragma unroll
      for (int q = 0; q < 4; ++q) dst[q] = src[q];
    } else {
      const uint4 z4 = make_uint4(0u, 0u, 0u, 0u);
#pragma unroll
      for (int q = 0; q < 4; ++q) dst[q] = z4;
    }
#endif
  };

  v8f acc[4];
#pragma unroll
  for (int t = 0; t < 4; ++t) acc[t] = (v8f){0.f, 0.f, 0.f, 0.f, 0.f, 0.f, 0.f, 0.f};

  // prologue: gather offset 0 into buf0; prefetch lookup for offset 1
  issue(0, lookup(0));
  int jnext = lookup(1);
#if HAVE_ASYNC_LDS
  wait_async0();
#endif
  __syncthreads();

  for (int k = 0; k < 125; ++k) {
    // issue next offset's gather into the other buffer (safe: last read of it
    // was at iteration k-1, fenced by the end-of-iteration barrier)
    if (k + 1 < 125) issue((k + 1) & 1, jnext);
    if (k + 2 < 125) jnext = lookup(k + 2);   // overlaps compute below

    // ---- 8 WMMAs on buffer k&1: 2 K-chunks x 4 N-tiles ----
    const __bf16* abase = &smem[k & 1][0];
#pragma unroll
    for (int c = 0; c < 2; ++c) {
      V16U A;
      const __bf16* arow = abase + rowInTile * 72 + c * 32 + hsel;
      A.h[0] = *(const v8bf*)(arow);        // K = base+0..7  (or +8..15 upper lanes)
      A.h[1] = *(const v8bf*)(arow + 16);   // K = base+16..23 (or +24..31)
      const __bf16* wb = Wp + ((size_t)(k * 2 + c) * 4) * 512 + lane * 16;
#pragma unroll
      for (int t = 0; t < 4; ++t) {
        v16bf B = *(const v16bf*)(wb + (size_t)t * 512);
        acc[t] = __builtin_amdgcn_wmma_f32_16x16x32_bf16(
            false, A.v, false, B, (short)0, acc[t], false, false);
      }
    }

#if HAVE_ASYNC_LDS
    wait_async0();      // own wave's async gather (k+1) complete
#endif
    __syncthreads();    // all waves: gather(k+1) visible, compute(k) reads done
  }

  // ---- store D per 16x16 f32 C/D layout; add bias ----
  const int mbase = blockIdx.x * 128 + wave * 16 + ((lane >= 16) ? 8 : 0);
#pragma unroll
  for (int t = 0; t < 4; ++t) {
    const int ch = t * 16 + (lane & 15);
    const float b = bias[ch];
#pragma unroll
    for (int jj = 0; jj < 8; ++jj) {
      const int m = mbase + jj;
      if (m < n) h[(size_t)m * 64 + ch] = acc[t][jj] + b;
    }
  }
}

// ---------------- BN statistics ----------------
__global__ void zero_stats_kernel(float* s) { s[threadIdx.x] = 0.f; }

__global__ void bn_reduce_kernel(const float* __restrict__ h, float* __restrict__ stats, int n) {
  const int c  = threadIdx.x & 63;
  const int rg = threadIdx.x >> 6;  // 0..3
  float s = 0.f, s2 = 0.f;
  for (int row = blockIdx.x * 4 + rg; row < n; row += gridDim.x * 4) {
    float v = h[(size_t)row * 64 + c];
    s += v; s2 += v * v;
  }
  __shared__ float sm[2][4][64];
  sm[0][rg][c] = s; sm[1][rg][c] = s2;
  __syncthreads();
  if (rg == 0) {
    s  = sm[0][0][c] + sm[0][1][c] + sm[0][2][c] + sm[0][3][c];
    s2 = sm[1][0][c] + sm[1][1][c] + sm[1][2][c] + sm[1][3][c];
    atomicAdd(&stats[c], s);
    atomicAdd(&stats[64 + c], s2);
  }
}

__global__ void bn_elu_kernel(const float* __restrict__ h, const float* __restrict__ stats,
                              const float* __restrict__ gamma, const float* __restrict__ beta,
                              float* __restrict__ out, int n) {
  int idx = blockIdx.x * 256 + threadIdx.x;
  if (idx >= n * 64) return;
  int c = idx & 63;
  float inv  = 1.0f / (float)n;
  float mean = stats[c] * inv;
  float var  = stats[64 + c] * inv - mean * mean;
  float x = (h[idx] - mean) * rsqrtf(var + 1e-5f) * gamma[c] + beta[c];
  out[idx] = (x > 0.f) ? x : (expf(x) - 1.f);
}

// ---------------- launcher ----------------
extern "C" void kernel_launch(void* const* d_in, const int* in_sizes, int n_in,
                              void* d_out, int out_size, void* d_ws, size_t ws_size,
                              hipStream_t stream) {
  const float* feats  = (const float*)d_in[0];
  const int*   coords = (const int*)  d_in[1];
  const float* Wt     = (const float*)d_in[2];
  const float* bias   = (const float*)d_in[3];
  const float* gamma  = (const float*)d_in[4];
  const float* beta   = (const float*)d_in[5];
  float* out = (float*)d_out;
  const int N = in_sizes[0] / 64;
  const int GRID = 128 * 128 * 128;

  // workspace layout
  char* ws = (char*)d_ws;
  int*    map    = (int*)ws;                                  // 8 MB
  size_t  o1     = (size_t)GRID * 4;
  __bf16* featsb = (__bf16*)(ws + o1);                        // N*64*2
  size_t  o2     = (o1 + (size_t)N * 64 * 2 + 31) & ~(size_t)31;
  __bf16* Wp     = (__bf16*)(ws + o2);                        // 1 MB packed weights
  size_t  o3     = (o2 + (size_t)125 * 4096 * 2 + 31) & ~(size_t)31;
  float*  h      = (float*)(ws + o3);                         // N*64*4
  size_t  o4     = (o3 + (size_t)N * 64 * 4 + 31) & ~(size_t)31;
  float*  stats  = (float*)(ws + o4);                         // 128 floats
  size_t  o5     = (o4 + 512 + 31) & ~(size_t)31;
  __bf16* zpage  = (__bf16*)(ws + o5);                        // 128-byte zero page

  fill_map_kernel<<<(GRID + 255) / 256, 256, 0, stream>>>(map, GRID);
  scatter_map_kernel<<<(N + 255) / 256, 256, 0, stream>>>(coords, map, N);
  zero_page_kernel<<<1, 32, 0, stream>>>((unsigned*)zpage);
  feats_to_bf16_kernel<<<((N * 64 / 8) + 255) / 256, 256, 0, stream>>>(feats, featsb, N * 64);
  pack_w_kernel<<<(125 * 4096 + 255) / 256, 256, 0, stream>>>(Wt, Wp);

  subm_conv_wmma_kernel<<<(N + 127) / 128, 256, 0, stream>>>(coords, map, featsb, Wp, bias,
                                                             zpage, h, N);

  zero_stats_kernel<<<1, 128, 0, stream>>>(stats);
  bn_reduce_kernel<<<512, 256, 0, stream>>>(h, stats, N);
  bn_elu_kernel<<<(N * 64 + 255) / 256, 256, 0, stream>>>(h, stats, gamma, beta, out, N);
}
